// GraphModelGAT_43654047596620
// MI455X (gfx1250) — compile-verified, matrix operand order checked by reading, not run on hardware
//
#include <hip/hip_runtime.h>
#include <hip/hip_bf16.h>

#define HD    256   // H*D (feature width, also D_IN)
#define HEADS 4
#define DH    64

typedef __attribute__((ext_vector_type(16))) __bf16 v16bf;
typedef __attribute__((ext_vector_type(8)))  float  v8f;

// ---- helpers -------------------------------------------------------------
__device__ __forceinline__ unsigned short f2bf(float f) {
  unsigned u = __float_as_uint(f);
  u += 0x7FFFu + ((u >> 16) & 1u);           // round to nearest even
  return (unsigned short)(u >> 16);
}
// monotonic float<->uint mapping so unsigned atomicMax == float max
__device__ __forceinline__ unsigned fflip(float f) {
  unsigned u = __float_as_uint(f);
  return (u & 0x80000000u) ? ~u : (u | 0x80000000u);
}
__device__ __forceinline__ float funflip(unsigned k) {
  unsigned u = (k & 0x80000000u) ? (k & 0x7FFFFFFFu) : ~k;
  return __uint_as_float(u);
}

// ---- conversion / packing ------------------------------------------------
__global__ void k_f32_to_bf16(const float* __restrict__ in,
                              unsigned short* __restrict__ out, int n) {
  int i = blockIdx.x * blockDim.x + threadIdx.x;
  if (i < n) out[i] = f2bf(in[i]);
}

// Pack W [256K x 256N] row-major into per-lane B fragments:
// Wpk[(((ks*16 + ct)*32 + lane)*16 + j], lane holds col ct*16+(lane&15),
// K halves at kb+0..7 and kb+16..23 with kb = ks*32 + (lane>>4)*8.
__global__ void k_pack_w(const float* __restrict__ W,
                         unsigned short* __restrict__ Wpk) {
  int id = blockIdx.x * blockDim.x + threadIdx.x;   // 0..65535
  int j    = id & 15;
  int lane = (id >> 4) & 31;
  int ct   = (id >> 9) & 15;
  int ks   = id >> 13;
  int n    = ct * 16 + (lane & 15);
  int kb   = ks * 32 + (lane >> 4) * 8;
  int K    = (j < 8) ? (kb + j) : (kb + 16 + (j - 8));
  Wpk[id] = f2bf(W[K * HD + n]);
}

// ---- WMMA GEMM: C[nrows,256] = A[nrows,256] @ W[256,256] (bf16 in, f32 out)
__global__ void __launch_bounds__(128) k_gemm_wmma(
    const unsigned short* __restrict__ Abf,
    const unsigned short* __restrict__ Wpk,
    float* __restrict__ C, int nrows) {
  const int lane = threadIdx.x & 31;
  const int wave = threadIdx.x >> 5;      // column group: 64 cols per wave
  const int hi   = lane >> 4;
  int row  = blockIdx.x * 16 + (lane & 15);
  int arow = row < nrows ? row : (nrows - 1);   // clamp loads, guard stores

  v8f acc0 = {}, acc1 = {}, acc2 = {}, acc3 = {};
  for (int ks = 0; ks < 8; ++ks) {
    int kb = ks * 32 + hi * 8;
    union { uint4 u[2]; v16bf v; } a;
    const uint4* ap = reinterpret_cast<const uint4*>(Abf + (size_t)arow * HD + kb);
    a.u[0] = ap[0];          // K = kb..kb+7
    a.u[1] = ap[2];          // K = kb+16..kb+23
    const uint4* bp = reinterpret_cast<const uint4*>(
        Wpk + ((size_t)((ks * 16 + wave * 4) * 32 + lane) << 4));
    union { uint4 u[2]; v16bf v; } b0, b1, b2, b3;
    b0.u[0] = bp[0];   b0.u[1] = bp[1];
    b1.u[0] = bp[64];  b1.u[1] = bp[65];
    b2.u[0] = bp[128]; b2.u[1] = bp[129];
    b3.u[0] = bp[192]; b3.u[1] = bp[193];
    acc0 = __builtin_amdgcn_wmma_f32_16x16x32_bf16(false, a.v, false, b0.v, (short)0, acc0, false, false);
    acc1 = __builtin_amdgcn_wmma_f32_16x16x32_bf16(false, a.v, false, b1.v, (short)0, acc1, false, false);
    acc2 = __builtin_amdgcn_wmma_f32_16x16x32_bf16(false, a.v, false, b2.v, (short)0, acc2, false, false);
    acc3 = __builtin_amdgcn_wmma_f32_16x16x32_bf16(false, a.v, false, b3.v, (short)0, acc3, false, false);
  }
  int colbase = wave * 64 + (lane & 15);
#pragma unroll
  for (int v = 0; v < 8; ++v) {
    int r = blockIdx.x * 16 + v + 8 * hi;   // C/D layout: VGPR v -> row v + 8*hi
    if (r < nrows) {
      float* crow = C + (size_t)r * HD + colbase;
      crow[0]  = acc0[v];
      crow[16] = acc1[v];
      crow[32] = acc2[v];
      crow[48] = acc3[v];
    }
  }
}

// ---- attention scores el/er ----------------------------------------------
__global__ void k_attn(const float* __restrict__ feat, const float* __restrict__ al,
                       const float* __restrict__ ar, float* __restrict__ el,
                       float* __restrict__ er, int n_nodes) {
  int id = blockIdx.x * blockDim.x + threadIdx.x;
  if (id >= n_nodes * HEADS) return;
  int nd = id >> 2, h = id & 3;
  const float* f  = feat + (size_t)nd * HD + h * DH;
  const float* a1 = al + h * DH;
  const float* a2 = ar + h * DH;
  float sl = 0.f, sr = 0.f;
#pragma unroll 8
  for (int d = 0; d < DH; ++d) { float fv = f[d]; sl += fv * a1[d]; sr += fv * a2[d]; }
  el[id] = sl; er[id] = sr;
}

__global__ void k_init(float* __restrict__ hout, unsigned* __restrict__ mu,
                       float* __restrict__ s, int n_nodes) {
  int i = blockIdx.x * blockDim.x + threadIdx.x;
  if (i < n_nodes * HD) hout[i] = 0.f;
  if (i < n_nodes * HEADS) { mu[i] = fflip(-__builtin_huge_valf()); s[i] = 0.f; }
}

// ---- edge softmax phases --------------------------------------------------
__global__ void k_edge1(const int* __restrict__ src, const int* __restrict__ dst,
                        const float* __restrict__ el, const float* __restrict__ er,
                        float* __restrict__ ebuf, unsigned* __restrict__ mu, int ne) {
  int id = blockIdx.x * blockDim.x + threadIdx.x;
  if (id >= ne * HEADS) return;
  int e = id >> 2, h = id & 3;
  float v = el[src[e] * HEADS + h] + er[dst[e] * HEADS + h];
  v = v > 0.f ? v : 0.2f * v;                 // leaky_relu(0.2)
  ebuf[id] = v;
  atomicMax(&mu[dst[e] * HEADS + h], fflip(v));
}

__global__ void k_edge2(const int* __restrict__ dst, float* __restrict__ ebuf,
                        const unsigned* __restrict__ mu, float* __restrict__ s, int ne) {
  int id = blockIdx.x * blockDim.x + threadIdx.x;
  if (id >= ne * HEADS) return;
  int e = id >> 2, h = id & 3;
  int dn = dst[e] * HEADS + h;
  float a = __expf(ebuf[id] - funflip(mu[dn]));
  ebuf[id] = a;
  atomicAdd(&s[dn], a);
}

// one 256-thread block per edge: coalesced 256-float gather + scatter-add
__global__ void __launch_bounds__(256) k_edge3(
    const int* __restrict__ src, const int* __restrict__ dst,
    const float* __restrict__ ebuf, const float* __restrict__ s,
    const float* __restrict__ feat, float* __restrict__ hout) {
  int e = blockIdx.x;
  int t = threadIdx.x;
  int h = t >> 6;
  int sn = src[e], dn = dst[e];
  float w = ebuf[e * HEADS + h] / s[dn * HEADS + h];
  atomicAdd(&hout[(size_t)dn * HD + t], w * feat[(size_t)sn * HD + t]);
}

__global__ void k_bias_relu(float* __restrict__ h, const float* __restrict__ b, int n_nodes) {
  int i = blockIdx.x * blockDim.x + threadIdx.x;
  if (i >= n_nodes * HD) return;
  float v = h[i] + b[i & (HD - 1)];
  h[i] = v > 0.f ? v : 0.f;
}

// ---- final dot scores -----------------------------------------------------
__global__ void k_score(const float* __restrict__ h, const int* __restrict__ u,
                        const int* __restrict__ v, float* __restrict__ out, int ne) {
  int id = blockIdx.x * blockDim.x + threadIdx.x;
  if (id >= ne * HEADS) return;
  int e = id >> 2, hh = id & 3;
  const float* pu = h + (size_t)u[e] * HD + hh * DH;
  const float* pv = h + (size_t)v[e] * HD + hh * DH;
  float sum = 0.f;
#pragma unroll 8
  for (int d = 0; d < DH; ++d) sum += pu[d] * pv[d];
  out[id] = sum;
}

// ---- orchestration --------------------------------------------------------
extern "C" void kernel_launch(void* const* d_in, const int* in_sizes, int n_in,
                              void* d_out, int out_size, void* d_ws, size_t ws_size,
                              hipStream_t stream) {
  const float* x   = (const float*)d_in[0];
  const int* src   = (const int*)d_in[1];
  const int* dst   = (const int*)d_in[2];
  const int* nsrc  = (const int*)d_in[3];
  const int* ndst  = (const int*)d_in[4];
  const int N = in_sizes[0] / HD;
  const int E = in_sizes[1];

  char* ws = (char*)d_ws;
  size_t off = 0;
  auto take = [&](size_t bytes) -> char* {
    char* p = ws + off; off = (off + bytes + 255) & ~(size_t)255; return p;
  };
  unsigned short* Abf = (unsigned short*)take((size_t)N * HD * 2);
  unsigned short* Wpk = (unsigned short*)take((size_t)HD * HD * 2);
  float*    F0   = (float*)take((size_t)N * HD * 4);    // per-node feat (pre-agg)
  float*    H0   = (float*)take((size_t)N * HD * 4);    // layer output
  float*    el   = (float*)take((size_t)N * HEADS * 4);
  float*    er   = (float*)take((size_t)N * HEADS * 4);
  unsigned* mu   = (unsigned*)take((size_t)N * HEADS * 4);
  float*    sbuf = (float*)take((size_t)N * HEADS * 4);
  float*    ebuf = (float*)take((size_t)E * HEADS * 4);

  const int TB = 256;
  int gNF   = (N * HD + TB - 1) / TB;
  int gNH   = (N * HEADS + TB - 1) / TB;
  int gEH   = (E * HEADS + TB - 1) / TB;
  int gGemm = (N + 15) / 16;

  for (int L = 0; L < 3; ++L) {
    const float* W  = (const float*)d_in[5 + 4 * L];
    const float* al = (const float*)d_in[6 + 4 * L];
    const float* ar = (const float*)d_in[7 + 4 * L];
    const float* bb = (const float*)d_in[8 + 4 * L];
    const float* inF = (L == 0) ? x : H0;

    k_f32_to_bf16<<<gNF, TB, 0, stream>>>(inF, Abf, N * HD);
    k_pack_w<<<(HD * HD) / TB, TB, 0, stream>>>(W, Wpk);
    k_gemm_wmma<<<gGemm, 128, 0, stream>>>(Abf, Wpk, F0, N);
    k_attn<<<gNH, TB, 0, stream>>>(F0, al, ar, el, er, N);
    k_init<<<gNF, TB, 0, stream>>>(H0, mu, sbuf, N);
    k_edge1<<<gEH, TB, 0, stream>>>(src, dst, el, er, ebuf, mu, E);
    k_edge2<<<gEH, TB, 0, stream>>>(dst, ebuf, mu, sbuf, E);
    k_edge3<<<E, 256, 0, stream>>>(src, dst, ebuf, sbuf, F0, H0);
    k_bias_relu<<<gNF, TB, 0, stream>>>(H0, bb, N);
  }

  float* out = (float*)d_out;
  k_score<<<gEH, TB, 0, stream>>>(H0, src, dst, out, E);
  k_score<<<gEH, TB, 0, stream>>>(H0, nsrc, ndst, out + (size_t)E * HEADS, E);
  (void)n_in; (void)out_size; (void)ws_size;
}
